// WindowAttention_50225347559867
// MI455X (gfx1250) — matrix operand order repaired and checked by reading, not compiled, not run on hardware
//
#include <hip/hip_runtime.h>
#include <hip/hip_bf16.h>

typedef __attribute__((ext_vector_type(16))) _Float16 v16h;
typedef __attribute__((ext_vector_type(8)))  float    v8f;

#define WIN_N    49
#define NP       64        // padded rows
#define DIM      128
#define NHEAD    4
#define HDIM     32
#define QKV_COLS 384
#define NWIN     64
#define SCALE    0.17677669529663687f  // 32^-0.5

// ---------------- fragment helpers (wave32 CDNA5 WMMA layouts) ----------------
// A: 16x32 f16, row-major source (stride in halves).
//   lane L: m = mBase + (L&15), hi = L>>4; VGPR p holds K = hi*16 + 2p (+1)
__device__ __forceinline__ v16h load_frag_a(const _Float16* base, int stride,
                                            int mBase, int kBase, int lane) {
  const int m  = mBase + (lane & 15);
  const int hi = lane >> 4;
  const unsigned int* p32 =
      reinterpret_cast<const unsigned int*>(base + m * stride + kBase);
  union { unsigned int u[8]; v16h v; } r;
#pragma unroll
  for (int p = 0; p < 8; ++p)
    r.u[p] = p32[p + hi * 8];
  return r.v;
}

// B: 32x16 f16 from COLUMN-major storage Bt[n][k] (stride in halves).
//   lane L: n = nBase + (L&15), hi = L>>4; VGPR p holds K = hi*16 + 2p (+1)
__device__ __forceinline__ v16h load_frag_b(const _Float16* base, int stride,
                                            int nBase, int kBase, int lane) {
  const int n  = nBase + (lane & 15);
  const int hi = lane >> 4;
  const unsigned int* p32 =
      reinterpret_cast<const unsigned int*>(base + n * stride + kBase + hi * 16);
  union { unsigned int u[8]; v16h v; } r;
#pragma unroll
  for (int p = 0; p < 8; ++p) r.u[p] = p32[p];
  return r.v;
}

__device__ __forceinline__ v8f wmma_f16(v16h a, v16h b, v8f c) {
  return __builtin_amdgcn_wmma_f32_16x16x32_f16(false, a, false, b,
                                                (short)0, c, false, false);
}

// ---------------- prep kernels (run once per launch) ----------------
// wt[n*128 + k] = (f16) qkv_w[k*384 + n]   (column-major f16, B-fragment ready)
__global__ void cvt_qkv_w(const float* __restrict__ w, _Float16* __restrict__ wt) {
  int i = blockIdx.x * 256 + threadIdx.x;
  if (i < QKV_COLS * DIM) {
    int n = i >> 7, k = i & 127;
    wt[i] = (_Float16)w[k * QKV_COLS + n];
  }
}
// pt[n*128 + k] = (f16) proj_w[k*128 + n]
__global__ void cvt_proj_w(const float* __restrict__ w, _Float16* __restrict__ pt) {
  int i = blockIdx.x * 256 + threadIdx.x;
  if (i < DIM * DIM) {
    int n = i >> 7, k = i & 127;
    pt[i] = (_Float16)w[k * DIM + n];
  }
}
// bm[m*49 + j] = bias_table[rel_index[m*49 + j]]
__global__ void gather_bias(const float* __restrict__ tab, const int* __restrict__ idx,
                            float* __restrict__ bm) {
  int i = blockIdx.x * 256 + threadIdx.x;
  if (i < WIN_N * WIN_N) bm[i] = tab[idx[i]];
}

// ---------------- fused window-attention kernel ----------------
// grid = 4096 windows, block = 128 threads (4 wave32; wave == head in attn phases)
__global__ __launch_bounds__(128) void
swin_window_attention(const float* __restrict__ x,       // [4096,49,128]
                      const float* __restrict__ mask,    // [64,49,49]
                      const _Float16* __restrict__ wt,   // [384,128] col-major f16
                      const float* __restrict__ qkv_b,   // [384]
                      const _Float16* __restrict__ pt,   // [128,128] col-major f16
                      const float* __restrict__ proj_b,  // [128]
                      const float* __restrict__ bm,      // [49,49]
                      float* __restrict__ out)           // [4096,49,128]
{
  // LDS: 128 KB, phase-aliased
  //   [0,64K)     sbuf f32[4][64][64]  (scores)   | first 16K alias xs f16[64][128]
  //   [64K,80K)   vt  f16[128][64]     (V^T, col-major B for PV)
  //   [80K,112K)  qs f16[64][128] + ks f16[64][128] | alias pbuf f16[4][64][64]
  //   [112K,128K) obuf f16[64][128] (PV result) | alias cmb f32[49*49] (bias+mask, phase<=3)
  __shared__ alignas(16) char smem[131072];
  float*    sbuf = reinterpret_cast<float*>(smem);
  _Float16* xs   = reinterpret_cast<_Float16*>(smem);
  _Float16* vt   = reinterpret_cast<_Float16*>(smem + 65536);
  _Float16* qs   = reinterpret_cast<_Float16*>(smem + 81920);
  _Float16* ks   = reinterpret_cast<_Float16*>(smem + 98304);
  _Float16* pbuf = reinterpret_cast<_Float16*>(smem + 81920);
  _Float16* obuf = reinterpret_cast<_Float16*>(smem + 114688);
  float*    cmb  = reinterpret_cast<float*>(smem + 114688);

  const int b    = blockIdx.x;
  const int tid  = threadIdx.x;
  const int wave = tid >> 5;
  const int lane = tid & 31;
  const int hi   = lane >> 4;
  const int nn   = lane & 15;
  const int widx = b & (NWIN - 1);

  // ---- phase 1: load x tile -> f16 LDS; stage bias+mask -> cmb ----
  {
    const float* xb = x + (size_t)b * WIN_N * DIM;
    for (int i = tid; i < NP * DIM; i += 128) {
      int row = i >> 7;
      xs[i] = (row < WIN_N) ? (_Float16)xb[i] : (_Float16)0.0f;
    }
    const float* mk = mask + (size_t)widx * WIN_N * WIN_N;
    for (int i = tid; i < WIN_N * WIN_N; i += 128)
      cmb[i] = bm[i] + mk[i];
  }
  __syncthreads();

  // ---- phase 2: QKV GEMM [64x128]@[128x384], route to qs/ks/vt ----
  // mt == wave (fixed per wave) => A fragments are loop-invariant (compiler hoists)
  for (int t = wave; t < 4 * 24; t += 4) {
    const int mt = t & 3, nt = t >> 2;   // mt: 0..3, nt: 0..23
    v8f c = {};
#pragma unroll
    for (int kk = 0; kk < 4; ++kk) {
      v16h a  = load_frag_a(xs, DIM, mt * 16, kk * 32, lane);
      v16h bf = load_frag_b(wt, DIM, nt * 16, kk * 32, lane);
      c = wmma_f16(a, bf, c);
    }
    const int col = nt * 16 + nn;
    const float bb = qkv_b[col];
    if (nt < 8) {            // Q (scaled)
#pragma unroll
      for (int r = 0; r < 8; ++r) {
        int m = mt * 16 + r + 8 * hi;
        qs[m * DIM + col] = (_Float16)((c[r] + bb) * SCALE);
      }
    } else if (nt < 16) {    // K
#pragma unroll
      for (int r = 0; r < 8; ++r) {
        int m = mt * 16 + r + 8 * hi;
        ks[m * DIM + (col - 128)] = (_Float16)(c[r] + bb);
      }
    } else {                 // V, stored transposed: vt[d_global][key]
#pragma unroll
      for (int r = 0; r < 8; ++r) {
        int m = mt * 16 + r + 8 * hi;
        vt[(col - 256) * NP + m] = (_Float16)(c[r] + bb);
      }
    }
  }
  __syncthreads();

  // ---- phase 3: scores = Q K^T + (bias+mask) (wave = head) ----
  {
    const int h = wave;
    v16h bk[4];
#pragma unroll
    for (int ntk = 0; ntk < 4; ++ntk)
      bk[ntk] = load_frag_b(ks, DIM, ntk * 16, h * HDIM, lane);  // K rows == col-major B
#pragma unroll
    for (int mt = 0; mt < 4; ++mt) {
      v16h a = load_frag_a(qs, DIM, mt * 16, h * HDIM, lane);
#pragma unroll
      for (int ntk = 0; ntk < 4; ++ntk) {
        v8f c = {};
        c = wmma_f16(a, bk[ntk], c);
        const int key = ntk * 16 + nn;
#pragma unroll
        for (int r = 0; r < 8; ++r) {
          int m = mt * 16 + r + 8 * hi;
          float s = c[r];
          if (key < WIN_N) {
            if (m < WIN_N) s += cmb[m * WIN_N + key];
          } else {
            s = -1e30f;   // padded keys -> zero probability
          }
          sbuf[h * (NP * NP) + m * NP + key] = s;
        }
      }
    }
  }
  __syncthreads();

  // ---- phase 4: row softmax (256 rows, 2 per thread) -> pbuf f16 ----
  for (int row = tid; row < NHEAD * NP; row += 128) {
    const int h = row >> 6, m = row & 63;
    float* srow = sbuf + h * (NP * NP) + m * NP;
    float mx = -3.4e38f;
#pragma unroll 8
    for (int j = 0; j < NP; ++j) mx = fmaxf(mx, srow[j]);
    float sum = 0.0f;
#pragma unroll 8
    for (int j = 0; j < NP; ++j) { float e = __expf(srow[j] - mx); srow[j] = e; sum += e; }
    const float inv = 1.0f / sum;
    _Float16* prow = pbuf + h * (NP * NP) + m * NP;
#pragma unroll 8
    for (int j = 0; j < NP; ++j) prow[j] = (_Float16)(srow[j] * inv);
  }
  __syncthreads();

  // ---- phase 5: out_h = P @ V   (wave = head) ----
  {
    const int h = wave;
    const _Float16* ph = pbuf + h * (NP * NP);
    v16h bv[2][2];
#pragma unroll
    for (int nt2 = 0; nt2 < 2; ++nt2)
#pragma unroll
      for (int kk = 0; kk < 2; ++kk)
        bv[nt2][kk] = load_frag_b(vt, NP, h * HDIM + nt2 * 16, kk * 32, lane);
#pragma unroll
    for (int mt = 0; mt < 4; ++mt) {
      v16h a0 = load_frag_a(ph, NP, mt * 16, 0, lane);
      v16h a1 = load_frag_a(ph, NP, mt * 16, 32, lane);
#pragma unroll
      for (int nt2 = 0; nt2 < 2; ++nt2) {
        v8f c = {};
        c = wmma_f16(a0, bv[nt2][0], c);
        c = wmma_f16(a1, bv[nt2][1], c);
        const int col = h * HDIM + nt2 * 16 + nn;
#pragma unroll
        for (int r = 0; r < 8; ++r) {
          int m = mt * 16 + r + 8 * hi;
          obuf[m * DIM + col] = (_Float16)c[r];
        }
      }
    }
  }
  __syncthreads();

  // ---- phase 6: proj GEMM [64x128]@[128x128] + bias -> global ----
  for (int t = wave; t < 32; t += 4) {
    const int mt = t & 3, nt = t >> 2;
    v8f c = {};
#pragma unroll
    for (int kk = 0; kk < 4; ++kk) {
      v16h a  = load_frag_a(obuf, DIM, mt * 16, kk * 32, lane);
      v16h bf = load_frag_b(pt, DIM, nt * 16, kk * 32, lane);
      c = wmma_f16(a, bf, c);
    }
    const int col = nt * 16 + nn;
    const float bb = proj_b[col];
#pragma unroll
    for (int r = 0; r < 8; ++r) {
      int m = mt * 16 + r + 8 * hi;
      if (m < WIN_N)
        out[((size_t)b * WIN_N + m) * DIM + col] = c[r] + bb;
    }
  }
}

extern "C" void kernel_launch(void* const* d_in, const int* in_sizes, int n_in,
                              void* d_out, int out_size, void* d_ws, size_t ws_size,
                              hipStream_t stream) {
  const float* x          = (const float*)d_in[0];
  const float* mask       = (const float*)d_in[1];
  const float* qkv_w      = (const float*)d_in[2];
  const float* qkv_b      = (const float*)d_in[3];
  const float* proj_w     = (const float*)d_in[4];
  const float* proj_b     = (const float*)d_in[5];
  const float* bias_table = (const float*)d_in[6];
  const int*   rel_index  = (const int*)d_in[7];
  float* out = (float*)d_out;

  // workspace: wt f16[384*128] | pt f16[128*128] | bm f32[49*49]  (~141 KB)
  _Float16* wt = (_Float16*)d_ws;
  _Float16* pt = wt + QKV_COLS * DIM;
  float*    bm = (float*)((char*)d_ws + (QKV_COLS * DIM + DIM * DIM) * sizeof(_Float16));

  cvt_qkv_w  <<<(QKV_COLS * DIM + 255) / 256, 256, 0, stream>>>(qkv_w, wt);
  cvt_proj_w <<<(DIM * DIM + 255) / 256,      256, 0, stream>>>(proj_w, pt);
  gather_bias<<<(WIN_N * WIN_N + 255) / 256,  256, 0, stream>>>(bias_table, rel_index, bm);

  swin_window_attention<<<4096, 128, 0, stream>>>(x, mask, wt, qkv_b, pt, proj_b, bm, out);
}
